// ComplexityAnalyzer_50053548867786
// MI455X (gfx1250) — matrix operand order repaired and checked by reading, not compiled, not run on hardware
//
#include <hip/hip_runtime.h>
#include <hip/hip_bf16.h>

typedef __attribute__((ext_vector_type(16))) _Float16 v16h;
typedef __attribute__((ext_vector_type(8)))  float    v8f;

#define HIST_BINS 64
#define BATCH     128
#define IMG_N     (512 * 512)          // elements per batch image
#define CHUNKS    32                   // blocks per batch image
#define CHUNK_F4  2048                 // float4s per chunk (8192 floats)

// ---------------------------------------------------------------------------
// Kernel 0: zero the per-batch histogram accumulators in workspace
// ---------------------------------------------------------------------------
__global__ void ca_zero_kernel(unsigned* __restrict__ p, int n) {
    int i = blockIdx.x * blockDim.x + threadIdx.x;
    if (i < n) p[i] = 0u;
}

// ---------------------------------------------------------------------------
// Kernel 1: bandwidth-bound histogram. 134 MB streamed once -> ~5.8us floor
// at 23.3 TB/s. Per-wave LDS sub-histograms cut DS-atomic contention.
// ---------------------------------------------------------------------------
__device__ __forceinline__ void ca_bin1(float v, unsigned* __restrict__ h) {
    if (v >= 0.0f && v <= 255.0f) {
        int i = (int)(v * (64.0f / 255.0f));   // floor for non-negative v
        i = (i > 63) ? 63 : i;                 // v == 255.0 -> last bin
        atomicAdd(&h[i], 1u);
    }
}

__global__ void ca_hist_kernel(const float* __restrict__ g,
                               unsigned* __restrict__ ghist) {
    __shared__ unsigned lh[8][HIST_BINS];      // one private hist per wave32
    const int tid  = threadIdx.x;
    const int wave = tid >> 5;

    for (int i = tid; i < 8 * HIST_BINS; i += 256)
        ((unsigned*)lh)[i] = 0u;
    __syncthreads();

    const int b     = blockIdx.y;
    const int chunk = blockIdx.x;
    const float4* __restrict__ src =
        reinterpret_cast<const float4*>(g + (long long)b * IMG_N) +
        (long long)chunk * CHUNK_F4;

    unsigned* __restrict__ h = lh[wave];
#pragma unroll
    for (int i = 0; i < 8; ++i) {              // 8 * 256 = 2048 float4 / block
        float4 v = src[i * 256 + tid];
        ca_bin1(v.x, h); ca_bin1(v.y, h); ca_bin1(v.z, h); ca_bin1(v.w, h);
    }
    __syncthreads();

    for (int bin = tid; bin < HIST_BINS; bin += 256) {
        unsigned s = 0;
#pragma unroll
        for (int w = 0; w < 8; ++w) s += lh[w][bin];
        atomicAdd(&ghist[b * HIST_BINS + bin], s);
    }
}

// ---------------------------------------------------------------------------
// Kernel 2: normalize + 2-layer MLP via V_WMMA_F32_16X16X32_F16.
// Single 256-thread block = 8 wave32s. ~0.8 MFLOP -> latency only.
//   L1: [128,64] x [64,32]  -> 8x2 tiles, K in 2 steps of 32 (4 WMMA / wave)
//   L2: [128,32] x [32,128] -> 8x8 tiles, K = 32        (8 WMMA / wave)
// ---------------------------------------------------------------------------
__global__ void ca_mlp_kernel(const unsigned* __restrict__ ghist,
                              const float* __restrict__ W1,
                              const float* __restrict__ b1,
                              const float* __restrict__ W2,
                              const float* __restrict__ b2,
                              float* __restrict__ out) {
    __shared__ _Float16 Hs[BATCH * HIST_BINS];   // 16 KB normalized hists
    __shared__ _Float16 W1s[32 * HIST_BINS];     //  4 KB  [n][k]
    __shared__ _Float16 W2s[128 * 32];           //  8 KB  [n][k]
    __shared__ _Float16 A1s[BATCH * 32];         //  8 KB  layer-1 activations

    const int tid = threadIdx.x;

    // --- L1-normalize each histogram row, convert to f16 ---
    for (int r = tid; r < BATCH; r += 256) {
        const unsigned* c = &ghist[r * HIST_BINS];
        float s = 0.0f;
#pragma unroll
        for (int k = 0; k < HIST_BINS; ++k) s += (float)c[k];
        float inv = 1.0f / fmaxf(s, 1e-12f);
#pragma unroll
        for (int k = 0; k < HIST_BINS; ++k)
            Hs[r * HIST_BINS + k] = (_Float16)((float)c[k] * inv);
    }
    // --- stage weights as f16 ---
    for (int i = tid; i < 32 * HIST_BINS; i += 256) W1s[i] = (_Float16)W1[i];
    for (int i = tid; i < 128 * 32;       i += 256) W2s[i] = (_Float16)W2[i];
    __syncthreads();

    const int wave = tid >> 5;          // 0..7 -> M-tile
    const int lane = tid & 31;
    const int mrow = lane & 15;         // A-frag row inside tile
    const int hi   = lane >> 4;         // lane group 0/1
    const int kb_a = hi * 8;            // A-frag K base per lane group
    const int ncol = lane & 15;         // B/C/D column
    const int kb_b = hi * 16;           // unused; B frags use kb_a (same k order as A)
    const int mt   = wave;

    // ---------------- Layer 1 ----------------
    v16h a0, a1;                         // A frags for K=[0,32) and [32,64)
#pragma unroll
    for (int v = 0; v < 8; ++v) {
        int k = (v >> 2) * 16 + kb_a + (v & 3) * 2;
        const _Float16* row = &Hs[(mt * 16 + mrow) * HIST_BINS];
        a0[2 * v]     = row[k];
        a0[2 * v + 1] = row[k + 1];
        a1[2 * v]     = row[32 + k];
        a1[2 * v + 1] = row[32 + k + 1];
    }
#pragma unroll
    for (int nt = 0; nt < 2; ++nt) {
        const int n = nt * 16 + ncol;
        v16h bb0, bb1;
#pragma unroll
        for (int v = 0; v < 8; ++v) {
            int k = (v >> 2) * 16 + kb_a + (v & 3) * 2;
            bb0[2 * v]     = W1s[n * HIST_BINS + k];
            bb0[2 * v + 1] = W1s[n * HIST_BINS + k + 1];
            bb1[2 * v]     = W1s[n * HIST_BINS + 32 + k];
            bb1[2 * v + 1] = W1s[n * HIST_BINS + 32 + k + 1];
        }
        v8f c = {};
        c = __builtin_amdgcn_wmma_f32_16x16x32_f16(false, a0, false, bb0,
                                                   (short)0, c, false, false);
        c = __builtin_amdgcn_wmma_f32_16x16x32_f16(false, a1, false, bb1,
                                                   (short)0, c, false, false);
        float bias = b1[n];
#pragma unroll
        for (int r = 0; r < 8; ++r) {
            float x = fmaxf(c[r] + bias, 0.0f);                  // bias + ReLU
            int m = mt * 16 + hi * 8 + r;
            A1s[m * 32 + n] = (_Float16)x;
        }
    }
    __syncthreads();

    // ---------------- Layer 2 ----------------
    v16h a2;
#pragma unroll
    for (int v = 0; v < 8; ++v) {
        int k = (v >> 2) * 16 + kb_a + (v & 3) * 2;
        const _Float16* row = &A1s[(mt * 16 + mrow) * 32];
        a2[2 * v]     = row[k];
        a2[2 * v + 1] = row[k + 1];
    }
#pragma unroll
    for (int nt = 0; nt < 8; ++nt) {
        const int n = nt * 16 + ncol;
        v16h bb;
#pragma unroll
        for (int v = 0; v < 8; ++v) {
            int k = (v >> 2) * 16 + kb_a + (v & 3) * 2;
            bb[2 * v]     = W2s[n * 32 + k];
            bb[2 * v + 1] = W2s[n * 32 + k + 1];
        }
        v8f c = {};
        c = __builtin_amdgcn_wmma_f32_16x16x32_f16(false, a2, false, bb,
                                                   (short)0, c, false, false);
        float bias = b2[n];
#pragma unroll
        for (int r = 0; r < 8; ++r) {
            int m = mt * 16 + hi * 8 + r;
            out[m * 128 + n] = c[r] + bias;
        }
    }
}

// ---------------------------------------------------------------------------
extern "C" void kernel_launch(void* const* d_in, const int* in_sizes, int n_in,
                              void* d_out, int out_size, void* d_ws, size_t ws_size,
                              hipStream_t stream) {
    (void)in_sizes; (void)n_in; (void)out_size; (void)ws_size;

    const float* grad = (const float*)d_in[0];
    const float* W1   = (const float*)d_in[1];
    const float* b1   = (const float*)d_in[2];
    const float* W2   = (const float*)d_in[3];
    const float* b2   = (const float*)d_in[4];
    float*    out  = (float*)d_out;
    unsigned* hist = (unsigned*)d_ws;            // BATCH * HIST_BINS u32

    const int nhist = BATCH * HIST_BINS;
    ca_zero_kernel<<<(nhist + 255) / 256, 256, 0, stream>>>(hist, nhist);
    ca_hist_kernel<<<dim3(CHUNKS, BATCH), 256, 0, stream>>>(grad, hist);
    ca_mlp_kernel<<<1, 256, 0, stream>>>(hist, W1, b1, W2, b2, out);
}
